// OjaUpdateVectorField_84859963834882
// MI455X (gfx1250) — matrix operand order repaired and checked by reading, not compiled, not run on hardware
//
#include <hip/hip_runtime.h>
#include <hip/hip_bf16.h>

// ---------------- Types for CDNA5 WMMA ----------------
typedef _Float16 f16;
typedef __attribute__((ext_vector_type(16))) _Float16 v16h;
typedef __attribute__((ext_vector_type(8)))  _Float16 v8h;
typedef __attribute__((ext_vector_type(4)))  _Float16 v4h;
typedef __attribute__((ext_vector_type(8)))  float    v8f;

#define BSZ      1024
#define IN_CH    512
#define HID      1024
#define N_HEADS  16
#define HEAD_DIM 64
#define KVB_OUT  (N_HEADS * (2 * HEAD_DIM + 1))   // 2064
#define LN_EPSF  1e-5f

// Padded LDS row strides (halves) so the 16 row-lanes map to distinct
// 4-bank groups: (stride_bytes/4) % 64 == 4.
#define A1_STRIDE 520    // GEMM1: K=512  -> 520*2  = 1040 B rows
#define A2_STRIDE 1032   // GEMM2: K=1024 -> 1032*2 = 2064 B rows

// ---------------------------------------------------------------------------
// Fragment loaders (wave32, 16-bit operands, 16x16x32 WMMA)
// A (16x32, MxK): lane L = row (L&15); elems 0..15 = K kbase+16*(L>>4).. +15.
// B (32x16, KxN): lane L = col (L&15); elems 0..15 = K kbase+16*(L>>4).. +15.
// ---------------------------------------------------------------------------
__device__ inline v16h load_frag_pair(const f16* p, int gap) {
  v8h lo = *(const v8h*)(p);
  v8h hi = *(const v8h*)(p + gap);
  v16h r;
#pragma unroll
  for (int i = 0; i < 8; ++i) { r[i] = lo[i]; r[i + 8] = hi[i]; }
  return r;
}
__device__ inline v16h load_a_frag(const f16* p) { return load_frag_pair(p, 8); }
__device__ inline v16h load_b_frag(const f16* p) { return load_frag_pair(p, 8); }

__device__ inline v8f wmma_f16(v16h a, v16h b, v8f c) {
  return __builtin_amdgcn_wmma_f32_16x16x32_f16(false, a, false, b,
                                                (short)0, c, false, false);
}

// ---------------------------------------------------------------------------
// Kernel 0: f32 -> f16 conversion (vectorized, n multiple of 4)
// ---------------------------------------------------------------------------
__global__ void cvt_f32_to_f16(const float* __restrict__ src,
                               f16* __restrict__ dst, int n4) {
  int i = blockIdx.x * blockDim.x + threadIdx.x;
  if (i < n4) {
    float4 v = ((const float4*)src)[i];
    v4h o;
    o[0] = (f16)v.x; o[1] = (f16)v.y; o[2] = (f16)v.z; o[3] = (f16)v.w;
    ((v4h*)dst)[i] = o;
  }
}

// ---------------------------------------------------------------------------
// Kernel 1: h = LayerNorm(x @ w_in^T + b_in) * gamma + beta, stored f16.
// 512 threads = 16 waves per block; block owns a 16-row slab.
// A slab (16x512 f16) staged once in LDS (shared by all waves).
// Each wave: 4 N-tiles (64 cols).  K-loop ping-pongs two B fragment sets
// (no buffer-rotation moves, no WAR on just-consumed B registers).
// ---------------------------------------------------------------------------
__global__ __launch_bounds__(512) void gemm1_ln_kernel(
    const f16* __restrict__ xh, const f16* __restrict__ w_inh,
    const float* __restrict__ b_in, const float* __restrict__ gamma,
    const float* __restrict__ beta, f16* __restrict__ h_lnh) {
  const int tid   = threadIdx.x;
  const int lane  = tid & 31;
  const int wave  = tid >> 5;           // 0..15
  const int mbase = blockIdx.x * 16;
  const int r16   = lane & 15;
  const int hl    = lane >> 4;

  __shared__ f16 aLDS[16 * A1_STRIDE];
  __shared__ float s_sum[16];
  __shared__ float s_sq[16];

  // ---- stage A slab: 16 rows x 512 halves; 512 threads x 16 halves ----
  {
    const int row = tid >> 5;           // 0..15
    const int ch  = (tid & 31) * 16;    // 0..496
    const f16* src = xh + (size_t)(mbase + row) * IN_CH + ch;
    f16* dst = aLDS + row * A1_STRIDE + ch;
    *(v8h*)(dst)     = *(const v8h*)(src);
    *(v8h*)(dst + 8) = *(const v8h*)(src + 8);
  }
  if (tid < 16) { s_sum[tid] = 0.0f; s_sq[tid] = 0.0f; }
  __syncthreads();

  v8f acc[4];
#pragma unroll
  for (int t = 0; t < 4; ++t) acc[t] = (v8f)0.0f;

  const int nb0 = wave * 64;            // wave covers cols [nb0, nb0+64)
  const f16* brow[4];
#pragma unroll
  for (int t = 0; t < 4; ++t)
    brow[t] = w_inh + (size_t)(nb0 + t * 16 + r16) * IN_CH + hl * 16;
  const f16* arow = aLDS + r16 * A1_STRIDE + hl * 16;

  // ---- ping-pong K loop (K = 512, step 64, last iteration peeled) ----
  v16h b0[4], b1[4];
#pragma unroll
  for (int t = 0; t < 4; ++t) b0[t] = load_b_frag(brow[t]);

  int kb = 0;
  for (; kb < IN_CH - 64; kb += 64) {
#pragma unroll
    for (int t = 0; t < 4; ++t) b1[t] = load_b_frag(brow[t] + kb + 32);
    {
      v16h a = load_a_frag(arow + kb);                 // ds_load
#pragma unroll
      for (int t = 0; t < 4; ++t) acc[t] = wmma_f16(a, b0[t], acc[t]);
    }
#pragma unroll
    for (int t = 0; t < 4; ++t) b0[t] = load_b_frag(brow[t] + kb + 64);
    {
      v16h a = load_a_frag(arow + kb + 32);            // ds_load
#pragma unroll
      for (int t = 0; t < 4; ++t) acc[t] = wmma_f16(a, b1[t], acc[t]);
    }
  }
  // tail: kb == IN_CH - 64
  {
#pragma unroll
    for (int t = 0; t < 4; ++t) b1[t] = load_b_frag(brow[t] + kb + 32);
    v16h a = load_a_frag(arow + kb);
#pragma unroll
    for (int t = 0; t < 4; ++t) acc[t] = wmma_f16(a, b0[t], acc[t]);
    v16h a2 = load_a_frag(arow + kb + 32);
#pragma unroll
    for (int t = 0; t < 4; ++t) acc[t] = wmma_f16(a2, b1[t], acc[t]);
  }

  // bias
#pragma unroll
  for (int t = 0; t < 4; ++t) {
    const float bias = b_in[nb0 + t * 16 + r16];
#pragma unroll
    for (int r = 0; r < 8; ++r) acc[t][r] += bias;
  }

  // ---- LayerNorm statistics over the full 1024-wide row ----
#pragma unroll
  for (int r = 0; r < 8; ++r) {
    float ps = 0.0f, pq = 0.0f;
#pragma unroll
    for (int t = 0; t < 4; ++t) { float v = acc[t][r]; ps += v; pq += v * v; }
#pragma unroll
    for (int m = 8; m >= 1; m >>= 1) {        // stays within 16-lane half
      ps += __shfl_xor(ps, m, 32);
      pq += __shfl_xor(pq, m, 32);
    }
    if (r16 == 0) {
      const int row = r + 8 * hl;
      atomicAdd(&s_sum[row], ps);             // ds_add_f32
      atomicAdd(&s_sq[row],  pq);
    }
  }
  __syncthreads();

  float mu[8], rstd[8];
#pragma unroll
  for (int r = 0; r < 8; ++r) {
    const int row = r + 8 * hl;
    const float m  = s_sum[row] * (1.0f / (float)HID);
    const float vv = s_sq[row]  * (1.0f / (float)HID) - m * m;
    mu[r]   = m;
    rstd[r] = rsqrtf(vv + LN_EPSF);
  }

#pragma unroll
  for (int t = 0; t < 4; ++t) {
    const int col = nb0 + t * 16 + r16;
    const float g  = gamma[col];
    const float be = beta[col];
#pragma unroll
    for (int r = 0; r < 8; ++r) {
      const int row = r + 8 * hl;
      const float v = (acc[t][r] - mu[r]) * rstd[r] * g + be;
      h_lnh[(size_t)(mbase + row) * HID + col] = (f16)v;
    }
  }
}

// ---------------------------------------------------------------------------
// Kernel 2: out2 = h_ln @ w_kvb^T + b_kvb  (1024 x 2064, K=1024), f32 out.
// 512 threads = 16 waves; block covers 256 columns; grid (64, 9).
// A slab (16x1024 f16) staged once in LDS; B ping-pong double-buffered.
// Inactive waves (col base >= 2064) clamp addresses and skip the store.
// ---------------------------------------------------------------------------
__global__ __launch_bounds__(512) void gemm2_kernel(
    const f16* __restrict__ h_lnh, const f16* __restrict__ w_kvbh,
    const float* __restrict__ b_kvb, float* __restrict__ out2) {
  const int tid   = threadIdx.x;
  const int lane  = tid & 31;
  const int wave  = tid >> 5;
  const int mbase = blockIdx.x * 16;
  const int r16   = lane & 15;
  const int hl    = lane >> 4;

  __shared__ f16 aLDS[16 * A2_STRIDE];

  // ---- stage A slab: 16 rows x 1024 halves; 512 threads x 32 halves ----
  {
    const int row = tid >> 5;
    const int ch  = (tid & 31) * 32;
    const f16* src = h_lnh + (size_t)(mbase + row) * HID + ch;
    f16* dst = aLDS + row * A2_STRIDE + ch;
#pragma unroll
    for (int q = 0; q < 4; ++q)
      *(v8h*)(dst + q * 8) = *(const v8h*)(src + q * 8);
  }
  __syncthreads();

  const int nb_raw  = blockIdx.y * 256 + wave * 16;
  const bool active = (nb_raw < KVB_OUT);
  const int nb      = active ? nb_raw : (KVB_OUT - 16);   // clamp: no OOB

  const f16* brow = w_kvbh + (size_t)(nb + r16) * HID + hl * 16;
  const f16* arow = aLDS + r16 * A2_STRIDE + hl * 16;

  // ---- ping-pong K loop (K = 1024, step 64, last iteration peeled) ----
  v8f acc = (v8f)0.0f;
  v16h b0 = load_b_frag(brow);
  v16h b1;
  int kb = 0;
  for (; kb < HID - 64; kb += 64) {
    b1 = load_b_frag(brow + kb + 32);
    acc = wmma_f16(load_a_frag(arow + kb), b0, acc);
    b0 = load_b_frag(brow + kb + 64);
    acc = wmma_f16(load_a_frag(arow + kb + 32), b1, acc);
  }
  // tail: kb == HID - 64
  b1  = load_b_frag(brow + kb + 32);
  acc = wmma_f16(load_a_frag(arow + kb), b0, acc);
  acc = wmma_f16(load_a_frag(arow + kb + 32), b1, acc);

  if (active) {
    const int col = nb + r16;
    const float bias = b_kvb[col];
#pragma unroll
    for (int r = 0; r < 8; ++r) {
      const int row = mbase + r + 8 * hl;
      out2[(size_t)row * KVB_OUT + col] = acc[r] + bias;
    }
  }
}

// ---------------------------------------------------------------------------
// Kernel 3 (bandwidth-critical): per (batch, head)
//   vs = tanh(raw[64:128]); lr = sigmoid(raw[128])
//   krem[j] = sum_i W[i][j] * vs[i]        (W read once, b128 loads)
//   ks = softmax(raw[0:64] - krem) * lr
//   dW[i][j] = vs[i] * ks[j]               (b128 stores)
// ---------------------------------------------------------------------------
__global__ __launch_bounds__(256) void head_update_kernel(
    const float* __restrict__ Wt, const float* __restrict__ out2,
    float* __restrict__ dW) {
  const int blk = blockIdx.x;                 // b*16 + head
  const int b   = blk >> 4;
  const int hd  = blk & 15;
  const float* raw = out2 + (size_t)b * KVB_OUT + hd * (2 * HEAD_DIM + 1);
  const float* Wb  = Wt + (size_t)blk * (HEAD_DIM * HEAD_DIM);
  float*       dWb = dW + (size_t)blk * (HEAD_DIM * HEAD_DIM);

  __shared__ __align__(16) float vsL[64];
  __shared__ __align__(16) float ksL[64];
  __shared__ float red[64][17];
  __shared__ float sred[64];
  __shared__ float lrS;

  const int t  = threadIdx.x;
  const int j0 = (t & 15) * 4;
  const int ig = t >> 4;

  // prefetch this thread's W cache lines so the 16KB tile read overlaps
  // the activation + barrier phase (global_prefetch_b8)
#pragma unroll
  for (int k = 0; k < 4; ++k)
    __builtin_prefetch(Wb + (size_t)(ig + 16 * k) * HEAD_DIM + j0, 0, 0);

  if (t < 64) {
    ksL[t] = raw[t];                       // raw k
    vsL[t] = tanhf(raw[HEAD_DIM + t]);     // vs
  }
  if (t == 0) lrS = 1.0f / (1.0f + expf(-raw[2 * HEAD_DIM]));
  __syncthreads();

  // ---- krem: thread t covers columns j0..j0+3, rows i = ig + 16k ----
  float p0 = 0.f, p1 = 0.f, p2 = 0.f, p3 = 0.f;
#pragma unroll
  for (int k = 0; k < 4; ++k) {
    const int i = ig + 16 * k;
    const float4 w = *(const float4*)(Wb + (size_t)i * HEAD_DIM + j0);
    const float v = vsL[i];
    p0 += w.x * v; p1 += w.y * v; p2 += w.z * v; p3 += w.w * v;
  }
  red[j0 + 0][ig] = p0; red[j0 + 1][ig] = p1;
  red[j0 + 2][ig] = p2; red[j0 + 3][ig] = p3;
  __syncthreads();

  float kk = 0.0f;
  if (t < 64) {
    float s = 0.0f;
#pragma unroll
    for (int g = 0; g < 16; ++g) s += red[t][g];
    kk = ksL[t] - s;
    sred[t] = kk;
  }
  __syncthreads();
  // ---- max over 64 ----
  if (t < 32) sred[t] = fmaxf(sred[t], sred[t + 32]); __syncthreads();
  if (t < 16) sred[t] = fmaxf(sred[t], sred[t + 16]); __syncthreads();
  if (t < 8)  sred[t] = fmaxf(sred[t], sred[t + 8]);  __syncthreads();
  if (t < 4)  sred[t] = fmaxf(sred[t], sred[t + 4]);  __syncthreads();
  if (t < 2)  sred[t] = fmaxf(sred[t], sred[t + 2]);  __syncthreads();
  if (t < 1)  sred[t] = fmaxf(sred[t], sred[t + 1]);  __syncthreads();
  const float mx = sred[0];
  __syncthreads();
  // ---- exp + sum over 64 ----
  float ex = 0.0f;
  if (t < 64) { ex = expf(kk - mx); sred[t] = ex; }
  __syncthreads();
  if (t < 32) sred[t] += sred[t + 32]; __syncthreads();
  if (t < 16) sred[t] += sred[t + 16]; __syncthreads();
  if (t < 8)  sred[t] += sred[t + 8];  __syncthreads();
  if (t < 4)  sred[t] += sred[t + 4];  __syncthreads();
  if (t < 2)  sred[t] += sred[t + 2];  __syncthreads();
  if (t < 1)  sred[t] += sred[t + 1];  __syncthreads();
  const float inv_sum = 1.0f / sred[0];
  __syncthreads();
  if (t < 64) ksL[t] = ex * inv_sum * lrS;
  __syncthreads();

  // ---- dW = vs ⊗ ks, b128 stores ----
#pragma unroll
  for (int k = 0; k < 4; ++k) {
    const int idx4 = t + 256 * k;        // float4 index, 1024 total
    const int i  = idx4 >> 4;
    const int jb = (idx4 & 15) * 4;
    const float v = vsL[i];
    const float4 kv = *(const float4*)(ksL + jb);
    float4 o;
    o.x = v * kv.x; o.y = v * kv.y; o.z = v * kv.z; o.w = v * kv.w;
    *(float4*)(dWb + (size_t)idx4 * 4) = o;
  }
}

// ---------------------------------------------------------------------------
// Launch
// ---------------------------------------------------------------------------
extern "C" void kernel_launch(void* const* d_in, const int* in_sizes, int n_in,
                              void* d_out, int out_size, void* d_ws, size_t ws_size,
                              hipStream_t stream) {
  const float* x      = (const float*)d_in[0];
  const float* W      = (const float*)d_in[1];
  const float* w_in   = (const float*)d_in[2];
  const float* b_in   = (const float*)d_in[3];
  const float* gamma  = (const float*)d_in[4];
  const float* beta   = (const float*)d_in[5];
  const float* w_kvb  = (const float*)d_in[6];
  const float* b_kvb  = (const float*)d_in[7];
  float* dW_out = (float*)d_out;

  // workspace layout (bytes)
  char* ws = (char*)d_ws;
  f16*   xh     = (f16*)(ws + 0);             // 1024*512*2   = 1048576
  f16*   w_inh  = (f16*)(ws + 524288);        // 1024*512*2   = 1048576
  f16*   w_kvbh = (f16*)(ws + 1048576);       // 2064*1024*2  = 4227072
  f16*   h_lnh  = (f16*)(ws + 5275648);       // 1024*1024*2  = 2097152
  float* out2   = (float*)(ws + 7372800);     // 1024*2064*4  = 8454144

  // 0) convert operands to f16
  {
    int n4;
    n4 = (BSZ * IN_CH) / 4;
    cvt_f32_to_f16<<<(n4 + 255) / 256, 256, 0, stream>>>(x, xh, n4);
    n4 = (HID * IN_CH) / 4;
    cvt_f32_to_f16<<<(n4 + 255) / 256, 256, 0, stream>>>(w_in, w_inh, n4);
    n4 = (KVB_OUT * HID) / 4;
    cvt_f32_to_f16<<<(n4 + 255) / 256, 256, 0, stream>>>(w_kvb, w_kvbh, n4);
  }

  // 1) fused GEMM1 + LayerNorm -> h_lnh (f16)
  gemm1_ln_kernel<<<BSZ / 16, 512, 0, stream>>>(xh, w_inh, b_in, gamma, beta, h_lnh);

  // 2) GEMM2 -> out2 (f32)
  gemm2_kernel<<<dim3(BSZ / 16, 9), 512, 0, stream>>>(h_lnh, w_kvbh, b_kvb, out2);

  // 3) per-(b,head) streaming update: W read once, dW written once
  head_update_kernel<<<BSZ * N_HEADS, 256, 0, stream>>>(W, out2, dW_out);
}
